// GFCA_49357764166330
// MI455X (gfx1250) — matrix operand order repaired and checked by reading, not compiled, not run on hardware
//
#include <hip/hip_runtime.h>

// ---------------------------------------------------------------------------
// CDNA5 (gfx1250) WMMA helpers: wave32, 16x16x32 bf16 -> f32 accumulate
// ---------------------------------------------------------------------------
typedef __attribute__((ext_vector_type(16))) __bf16 v16bf;
typedef __attribute__((ext_vector_type(8)))  __bf16 v8bf;
typedef __attribute__((ext_vector_type(8)))  float  v8f;

static __device__ __forceinline__ v8f wmma_bf16(v16bf a, v16bf b, v8f c) {
    // 8 args: (neg_a, A, neg_b, B, c_mod, C, reuse_a, reuse_b)
    return __builtin_amdgcn_wmma_f32_16x16x32_bf16(false, a, false, b, (short)0, c,
                                                   false, false);
}

// A fragment: 16x32 (MxK) bf16 from LDS, row stride `stride` elements.
// Both 8-element groups are contiguous -> 2x ds_load_b128.
static __device__ __forceinline__ v16bf frag_a(const __bf16* lds, int stride, int lane) {
    const int half = (lane >> 4) & 1;
    const int m    = lane & 15;
    const __bf16* p = lds + m * stride + half * 16;
    v8bf lo = *(const v8bf*)p;          // K = half*16 + 0..7
    v8bf hi = *(const v8bf*)(p + 8);    // K = half*16 + 8..15
    return __builtin_shufflevector(lo, hi, 0, 1, 2, 3, 4, 5, 6, 7,
                                           8, 9, 10, 11, 12, 13, 14, 15);
}

// Swizzled B panel (N=64): K2 x 64 weight panel stored fragment-ready as
// [kt][ct][lane][j] so each lane's 16 B elements are contiguous.
static __device__ __forceinline__ int bswz(int k, int n) {
    const int kt   = k >> 5;
    const int lane = (((k >> 4) & 1) << 4) | (n & 15);
    const int ct   = n >> 4;
    return (((kt * 4 + ct) * 32) + lane) * 16 + (k & 15);
}

static __device__ __forceinline__ v16bf frag_b(const __bf16* panel, int kt, int ct, int lane) {
    const __bf16* p = panel + (((kt * 4 + ct) * 32) + lane) * 16;
    v8bf lo = *(const v8bf*)p;
    v8bf hi = *(const v8bf*)(p + 8);
    return __builtin_shufflevector(lo, hi, 0, 1, 2, 3, 4, 5, 6, 7,
                                           8, 9, 10, 11, 12, 13, 14, 15);
}

// Narrow swizzled B panel (N=16): [kt][lane][j]
static __device__ __forceinline__ v16bf frag_b16(const __bf16* panel, int kt, int lane) {
    const __bf16* p = panel + (kt * 32 + lane) * 16;
    v8bf lo = *(const v8bf*)p;
    v8bf hi = *(const v8bf*)(p + 8);
    return __builtin_shufflevector(lo, hi, 0, 1, 2, 3, 4, 5, 6, 7,
                                           8, 9, 10, 11, 12, 13, 14, 15);
}

static __device__ __forceinline__ void atomicMaxF(float* addr, float val) {
    // monotone int trick (valid with -inf init)
    if (val >= 0.0f) atomicMax((int*)addr, __float_as_int(val));
    else             atomicMin((unsigned int*)addr, (unsigned int)__float_as_int(val));
}

// ---------------------------------------------------------------------------
// Generic GEMM (compile-time K): C[M x 64] = act(A[M x K] @ W[K x 64] + bias).
// Block = 128 threads (4 waves); 16 rows/block; wave w -> cols [16w,16w+16).
// ---------------------------------------------------------------------------
template <int K, bool RELU>
__global__ void __launch_bounds__(128)
gemm_n64_t(const float* __restrict__ A, const float* __restrict__ W,
           const float* __restrict__ bias, float* __restrict__ C, int M) {
    constexpr int K2 = ((K + 31) / 32) * 32;
    __shared__ __align__(16) __bf16 sA[16 * K2];
    __shared__ __align__(16) __bf16 sW[K2 * 64];
    const int tid  = threadIdx.x;
    const int lane = tid & 31;
    const int wave = tid >> 5;
    const int row0 = blockIdx.x * 16;

    for (int i = tid; i < 16 * K2; i += 128) {
        const int r = i / K2, c = i - r * K2;
        float v = 0.0f;
        if (c < K && (row0 + r) < M) v = A[(size_t)(row0 + r) * K + c];
        sA[i] = (__bf16)v;
    }
    for (int i = tid; i < K2 * 64; i += 128) {
        const int k = i >> 6, n = i & 63;
        sW[bswz(k, n)] = (__bf16)((k < K) ? W[(size_t)k * 64 + n] : 0.0f);
    }
    __syncthreads();

    v8f acc = {};
#pragma unroll
    for (int kt = 0; kt < K2 / 32; ++kt) {
        v16bf a = frag_a(sA + kt * 32, K2, lane);
        v16bf b = frag_b(sW, kt, wave, lane);
        acc = wmma_bf16(a, b, acc);
    }

    const int n  = wave * 16 + (lane & 15);
    const int mh = ((lane >> 4) & 1) * 8;
    const float bv = bias[n];
    if (row0 + 16 <= M) {
#pragma unroll
        for (int r = 0; r < 8; ++r) {
            float v = acc[r] + bv;
            if (RELU) v = fmaxf(v, 0.0f);
            C[(size_t)(row0 + r + mh) * 64 + n] = v;
        }
    } else {
#pragma unroll
        for (int r = 0; r < 8; ++r) {
            const int row = row0 + r + mh;
            if (row < M) {
                float v = acc[r] + bv;
                if (RELU) v = fmaxf(v, 0.0f);
                C[(size_t)row * 64 + n] = v;
            }
        }
    }
}

// ---------------------------------------------------------------------------
// Fused Q/K/V/skip GEMM: one 16x64 A tile of h, four 64x64 swizzled weights.
// ---------------------------------------------------------------------------
__global__ void __launch_bounds__(128)
gemm_qkvs(const float* __restrict__ h,
          const float* __restrict__ Wq, const float* __restrict__ bq,
          const float* __restrict__ Wk, const float* __restrict__ bk,
          const float* __restrict__ Wv, const float* __restrict__ bv,
          const float* __restrict__ Ws, const float* __restrict__ bs,
          float* __restrict__ q, float* __restrict__ k, float* __restrict__ v,
          float* __restrict__ skip, int M) {
    __shared__ __align__(16) __bf16 sA[16 * 64];
    __shared__ __align__(16) __bf16 sW[4][64 * 64];
    __shared__ float sB[4][64];
    const int tid  = threadIdx.x;
    const int lane = tid & 31;
    const int wave = tid >> 5;
    const int row0 = blockIdx.x * 16;

    const float* Wp[4] = {Wq, Wk, Wv, Ws};
    const float* bp[4] = {bq, bk, bv, bs};
    float*       op[4] = {q, k, v, skip};

    for (int i = tid; i < 16 * 64; i += 128) {
        const int r = i >> 6, c = i & 63;
        float val = 0.0f;
        if ((row0 + r) < M) val = h[(size_t)(row0 + r) * 64 + c];
        sA[i] = (__bf16)val;
    }
#pragma unroll
    for (int w = 0; w < 4; ++w) {
        for (int i = tid; i < 64 * 64; i += 128) {
            const int kq = i >> 6, n = i & 63;
            sW[w][bswz(kq, n)] = (__bf16)Wp[w][i];
        }
        if (tid < 64) sB[w][tid] = bp[w][tid];
    }
    __syncthreads();

    const int n  = wave * 16 + (lane & 15);
    const int mh = ((lane >> 4) & 1) * 8;
    const bool full = (row0 + 16) <= M;
#pragma unroll
    for (int w = 0; w < 4; ++w) {
        v8f acc = {};
#pragma unroll
        for (int kt = 0; kt < 2; ++kt) {
            v16bf a = frag_a(sA + kt * 32, 64, lane);
            v16bf b = frag_b(sW[w], kt, wave, lane);
            acc = wmma_bf16(a, b, acc);
        }
        const float bval = sB[w][n];
        if (full) {
#pragma unroll
            for (int r = 0; r < 8; ++r)
                op[w][(size_t)(row0 + r + mh) * 64 + n] = acc[r] + bval;
        } else {
#pragma unroll
            for (int r = 0; r < 8; ++r) {
                const int row = row0 + r + mh;
                if (row < M) op[w][(size_t)row * 64 + n] = acc[r] + bval;
            }
        }
    }
}

// ---------------------------------------------------------------------------
// Init: segmax = -inf, denom = 0, agg = 0
// ---------------------------------------------------------------------------
__global__ void init_buffers(float* __restrict__ segmax, float* __restrict__ denom,
                             float* __restrict__ agg, int N) {
    const int i = blockIdx.x * blockDim.x + threadIdx.x;
    if (i < N * 64) {
        agg[i] = 0.0f;
        if (i < N * 2) { segmax[i] = -__builtin_inff(); denom[i] = 0.0f; }
    }
}

// ---------------------------------------------------------------------------
// Edge pass 1 (all-WMMA):
//   e = cos(rel_t * w_t + b_t) @ We + be            (WMMA, folded into k_j)
//   s1 = relu([q_i | k_j] @ A1[h] + a1)             (WMMA, 8 cols zero-padded)
//   score = s1 @ A2[h] + a2 ; atomic segment-max over dst
// Block = 128 threads = 128 edges; wave w owns edge rows [32w, 32w+32).
// ---------------------------------------------------------------------------
__global__ void __launch_bounds__(128)
edge_score(const int* __restrict__ ei, const float* __restrict__ node_time,
           const float* __restrict__ edge_time,
           const float* __restrict__ w_t, const float* __restrict__ b_t,
           const float* __restrict__ We, const float* __restrict__ be,
           const float* __restrict__ A1, const float* __restrict__ a1,
           const float* __restrict__ A2, const float* __restrict__ a2,
           const float* __restrict__ q, const float* __restrict__ k,
           float* __restrict__ score, float* __restrict__ segmax, int E) {
    __shared__ __align__(16) __bf16 sQ[128 * 64];     // gathered q[dst]
    __shared__ __align__(16) __bf16 sK[128 * 64];     // gathered k[src] (+e+be)
    __shared__ __align__(16) __bf16 sEnc[128 * 32];
    __shared__ __align__(16) __bf16 sWe[32 * 64];
    __shared__ __align__(16) __bf16 sB1[2][2 * 32 * 16]; // A1[h] 64x16 panels
    __shared__ float sS1[128 * 16];                    // one head's s1 tile
    __shared__ int   sSrc[128], sDst[128];
    __shared__ float sa1[16], sA2v[16], sa2[2], sBe[64];

    const int tid  = threadIdx.x;
    const int lane = tid & 31;
    const int wave = tid >> 5;
    const int e    = blockIdx.x * 128 + tid;
    const int ec   = (e < E) ? e : (E - 1);

    // Stage We (N=64 swizzle) and A1 head panels (N=16 swizzle, cols 8..15 = 0)
    for (int i = tid; i < 32 * 64; i += 128) {
        const int kq = i >> 6, n = i & 63;
        sWe[bswz(kq, n)] = (__bf16)We[i];
    }
    for (int i = tid; i < 2 * 64 * 16; i += 128) {
        const int hh = i >> 10;
        const int rem = i & 1023;
        const int kq = rem >> 4, n = rem & 15;
        const float val = (n < 8) ? A1[(hh * 64 + kq) * 8 + n] : 0.0f;
        const int kt = kq >> 5, kr = kq & 31;
        sB1[hh][((kt * 32) + ((kr >> 4) * 16 + n)) * 16 + (kr & 15)] = (__bf16)val;
    }
    if (tid < 16) { sa1[tid] = a1[tid]; sA2v[tid] = A2[tid]; }
    if (tid < 2)  sa2[tid] = a2[tid];
    if (tid < 64) sBe[tid] = be[tid];

    const int src = ei[ec];
    const int dst = ei[E + ec];
    sSrc[tid] = src;
    sDst[tid] = dst;
    const float rel_t = node_time[src] - edge_time[ec];
#pragma unroll
    for (int j = 0; j < 32; ++j)
        sEnc[tid * 32 + j] = (__bf16)cosf(rel_t * w_t[j] + b_t[j]);
    __syncthreads();

    // Cooperative gathers of q[dst] and k[src] rows into bf16 LDS tiles.
    for (int idx = tid; idx < 128 * 64; idx += 128) {
        const int r = idx >> 6, c = idx & 63;
        sQ[idx] = (__bf16)q[(size_t)sDst[r] * 64 + c];
        sK[idx] = (__bf16)k[(size_t)sSrc[r] * 64 + c];
    }
    __syncthreads();

    // e-matrix WMMA, folded into sK: k_j = k[src] + e + be
    const int cn  = lane & 15;
    const int mh  = ((lane >> 4) & 1) * 8;
#pragma unroll
    for (int rt = 0; rt < 2; ++rt) {
        const int r0 = wave * 32 + rt * 16;
        v16bf a = frag_a(sEnc + r0 * 32, 32, lane);
#pragma unroll
        for (int ct = 0; ct < 4; ++ct) {
            v8f acc = {};
            v16bf b = frag_b(sWe, 0, ct, lane);
            acc = wmma_bf16(a, b, acc);
            const int n = ct * 16 + cn;
#pragma unroll
            for (int r = 0; r < 8; ++r) {
                const int idx = (r0 + r + mh) * 64 + n;
                sK[idx] = (__bf16)((float)sK[idx] + acc[r] + sBe[n]);
            }
        }
    }
    __syncthreads();

    // Per-head score MLP via WMMA: s1 = q_i @ A1[:C] + k_j @ A1[C:]
#pragma unroll
    for (int hh = 0; hh < 2; ++hh) {
#pragma unroll
        for (int rt = 0; rt < 2; ++rt) {
            const int r0 = wave * 32 + rt * 16;
            v8f acc = {};
            v16bf aq = frag_a(sQ + r0 * 64 + hh * 32, 64, lane);
            v16bf b0 = frag_b16(sB1[hh], 0, lane);
            acc = wmma_bf16(aq, b0, acc);
            v16bf ak = frag_a(sK + r0 * 64 + hh * 32, 64, lane);
            v16bf b1 = frag_b16(sB1[hh], 1, lane);
            acc = wmma_bf16(ak, b1, acc);
#pragma unroll
            for (int r = 0; r < 8; ++r)
                sS1[(r0 + r + mh) * 16 + cn] = acc[r];
        }
        __syncthreads();
        if (e < E) {
            float sc = sa2[hh];
#pragma unroll
            for (int m = 0; m < 8; ++m)
                sc += fmaxf(sS1[tid * 16 + m] + sa1[hh * 8 + m], 0.0f) * sA2v[hh * 8 + m];
            score[(size_t)e * 2 + hh] = sc;
            atomicMaxF(&segmax[(size_t)dst * 2 + hh], sc);
        }
        __syncthreads();
    }
}

// ---------------------------------------------------------------------------
// Edge pass 2 (elementwise over E*H): ex = exp(score - max[dst]); denom += ex
// ---------------------------------------------------------------------------
__global__ void seg_softmax_norm(const int* __restrict__ ei, float* __restrict__ score,
                                 const float* __restrict__ segmax,
                                 float* __restrict__ denom, int E) {
    const int idx = blockIdx.x * blockDim.x + threadIdx.x;
    if (idx >= 2 * E) return;
    const int e = idx >> 1, hh = idx & 1;
    const int dst = ei[E + e];
    const float ex = expf(score[idx] - segmax[dst * 2 + hh]);
    score[idx] = ex;
    atomicAdd(&denom[dst * 2 + hh], ex);
}

// ---------------------------------------------------------------------------
// Edge pass 3: recompute e via WMMA; agg[dst] += alpha * (v[src] + e)
// ---------------------------------------------------------------------------
__global__ void __launch_bounds__(128)
edge_agg(const int* __restrict__ ei, const float* __restrict__ node_time,
         const float* __restrict__ edge_time,
         const float* __restrict__ w_t, const float* __restrict__ b_t,
         const float* __restrict__ We, const float* __restrict__ be,
         const float* __restrict__ v, const float* __restrict__ exbuf,
         const float* __restrict__ denom, float* __restrict__ agg, int E) {
    __shared__ __align__(16) __bf16 sEnc[128 * 32];
    __shared__ __align__(16) __bf16 sWe[32 * 64];
    __shared__ float  sE[128 * 64];
    __shared__ float  sBe[64];

    const int tid  = threadIdx.x;
    const int lane = tid & 31;
    const int wave = tid >> 5;
    const int e    = blockIdx.x * 128 + tid;
    const int ec   = (e < E) ? e : (E - 1);

    for (int i = tid; i < 32 * 64; i += 128) {
        const int kq = i >> 6, n = i & 63;
        sWe[bswz(kq, n)] = (__bf16)We[i];
    }
    if (tid < 64) sBe[tid] = be[tid];

    const int src = ei[ec];
    const int dst = ei[E + ec];
    const float rel_t = node_time[src] - edge_time[ec];
#pragma unroll
    for (int j = 0; j < 32; ++j)
        sEnc[tid * 32 + j] = (__bf16)cosf(rel_t * w_t[j] + b_t[j]);
    __syncthreads();

#pragma unroll
    for (int rt = 0; rt < 2; ++rt) {
        const int r0 = wave * 32 + rt * 16;
        v16bf a = frag_a(sEnc + r0 * 32, 32, lane);
#pragma unroll
        for (int ct = 0; ct < 4; ++ct) {
            v8f acc = {};
            v16bf b = frag_b(sWe, 0, ct, lane);
            acc = wmma_bf16(a, b, acc);
            const int n  = ct * 16 + (lane & 15);
            const int mh = ((lane >> 4) & 1) * 8;
#pragma unroll
            for (int r = 0; r < 8; ++r)
                sE[(r0 + r + mh) * 64 + n] = acc[r] + sBe[n];
        }
    }
    __syncthreads();

    if (e < E) {
        const float* vrow = v + (size_t)src * 64;
        __builtin_prefetch(vrow, 0, 0);
#pragma unroll
        for (int hh = 0; hh < 2; ++hh) {
            const float alpha = exbuf[(size_t)e * 2 + hh]
                              / (denom[(size_t)dst * 2 + hh] + 1e-16f);
            for (int c = 0; c < 32; ++c) {
                const float vv = vrow[hh * 32 + c] + sE[tid * 64 + hh * 32 + c];
                atomicAdd(&agg[(size_t)dst * 64 + hh * 32 + c], alpha * vv);
            }
        }
    }
}

// ---------------------------------------------------------------------------
// Node fusion: h1pre = agg + skip; encodings; attention fusion; -> cc[N x 72]
// ---------------------------------------------------------------------------
__global__ void node_fuse(const float* __restrict__ agg, const float* __restrict__ skip,
                          const float* __restrict__ tfi, const float* __restrict__ degv,
                          const float* __restrict__ w_tf, const float* __restrict__ b_tf,
                          const float* __restrict__ w_deg, const float* __restrict__ b_deg,
                          const float* __restrict__ Wenc, const float* __restrict__ benc,
                          const float* __restrict__ Wx, const float* __restrict__ bx,
                          float* __restrict__ cc, int N) {
    const int n = blockIdx.x * blockDim.x + threadIdx.x;
    if (n >= N) return;
    float xp[8];
#pragma unroll
    for (int j = 0; j < 8; ++j) xp[j] = bx[j];
    for (int c = 0; c < 64; ++c) {
        const float hv = agg[(size_t)n * 64 + c] + skip[(size_t)n * 64 + c];
        cc[(size_t)n * 72 + c] = hv;
#pragma unroll
        for (int j = 0; j < 8; ++j) xp[j] += hv * Wx[c * 8 + j];
    }
#pragma unroll
    for (int j = 0; j < 8; ++j) xp[j] = tanhf(xp[j]);

    const float tfv = tfi[n], dgv = degv[n];
    float e0[8], e1[8];
#pragma unroll
    for (int j = 0; j < 8; ++j) {
        e0[j] = tfv * w_tf[j] + b_tf[j];
        e1[j] = dgv * w_deg[j] + b_deg[j];
    }
    float s0 = 0.0f, s1 = 0.0f;
#pragma unroll
    for (int j = 0; j < 8; ++j) {
        float p0 = benc[j], p1 = benc[j];
#pragma unroll
        for (int i = 0; i < 8; ++i) {
            p0 += e0[i] * Wenc[i * 8 + j];
            p1 += e1[i] * Wenc[i * 8 + j];
        }
        s0 += tanhf(p0) * xp[j];
        s1 += tanhf(p1) * xp[j];
    }
    const float mx = fmaxf(s0, s1);
    const float x0 = expf(s0 - mx), x1 = expf(s1 - mx);
    const float inv = 1.0f / (x0 + x1);
    const float w0 = x0 * inv, w1 = x1 * inv;
#pragma unroll
    for (int j = 0; j < 8; ++j)
        cc[(size_t)n * 72 + 64 + j] = e0[j] * w0 + e1[j] * w1;
}

// ---------------------------------------------------------------------------
// Output head: logits = h1 @ Wout + bout; log_softmax; pass-through y, mask.
// ---------------------------------------------------------------------------
__global__ void out_head(const float* __restrict__ h1, const float* __restrict__ Wout,
                         const float* __restrict__ bout, const int* __restrict__ y,
                         const unsigned char* __restrict__ mask,
                         float* __restrict__ out_ls, float* __restrict__ out_y,
                         float* __restrict__ out_m, int N) {
    const int n = blockIdx.x * blockDim.x + threadIdx.x;
    if (n >= N) return;
    float l0 = bout[0], l1 = bout[1];
    for (int c = 0; c < 64; ++c) {
        const float hv = h1[(size_t)n * 64 + c];
        l0 += hv * Wout[c * 2 + 0];
        l1 += hv * Wout[c * 2 + 1];
    }
    const float mx  = fmaxf(l0, l1);
    const float lse = mx + logf(expf(l0 - mx) + expf(l1 - mx));
    out_ls[(size_t)n * 2 + 0] = l0 - lse;
    out_ls[(size_t)n * 2 + 1] = l1 - lse;
    out_y[n] = (float)y[n];
    out_m[n] = mask[n] ? 1.0f : 0.0f;
}

// ---------------------------------------------------------------------------
// Host launcher
// ---------------------------------------------------------------------------
extern "C" void kernel_launch(void* const* d_in, const int* in_sizes, int n_in,
                              void* d_out, int out_size, void* d_ws, size_t ws_size,
                              hipStream_t stream) {
    (void)n_in; (void)out_size; (void)ws_size;
    const float* x         = (const float*)d_in[0];
    const float* node_time = (const float*)d_in[1];
    const float* edge_time = (const float*)d_in[2];
    const float* tfi       = (const float*)d_in[3];
    const float* degv      = (const float*)d_in[4];
    const float* w_t   = (const float*)d_in[5];
    const float* b_t   = (const float*)d_in[6];
    const float* W_lin = (const float*)d_in[7];
    const float* b_lin = (const float*)d_in[8];
    const float* Wq    = (const float*)d_in[9];
    const float* bq    = (const float*)d_in[10];
    const float* Wk    = (const float*)d_in[11];
    const float* bk    = (const float*)d_in[12];
    const float* Wv    = (const float*)d_in[13];
    const float* bv    = (const float*)d_in[14];
    const float* We    = (const float*)d_in[15];
    const float* be    = (const float*)d_in[16];
    const float* A1    = (const float*)d_in[17];
    const float* a1    = (const float*)d_in[18];
    const float* A2    = (const float*)d_in[19];
    const float* a2    = (const float*)d_in[20];
    const float* Wskip = (const float*)d_in[21];
    const float* bskip = (const float*)d_in[22];
    const float* w_tf  = (const float*)d_in[23];
    const float* b_tf  = (const float*)d_in[24];
    const float* w_deg = (const float*)d_in[25];
    const float* b_deg = (const float*)d_in[26];
    const float* Wenc  = (const float*)d_in[27];
    const float* benc  = (const float*)d_in[28];
    const float* Wx    = (const float*)d_in[29];
    const float* bx    = (const float*)d_in[30];
    const float* Wcomb = (const float*)d_in[31];
    const float* bcomb = (const float*)d_in[32];
    const float* Wout  = (const float*)d_in[33];
    const float* bout  = (const float*)d_in[34];
    const int* ei = (const int*)d_in[35];
    const int* y  = (const int*)d_in[36];
    const unsigned char* mask = (const unsigned char*)d_in[37];

    const int N = in_sizes[1];   // node_time
    const int E = in_sizes[2];   // edge_time (E x 1)

    // Workspace carve-out (floats)
    float* ws = (float*)d_ws;
    size_t off = 0;
    float* h      = ws + off; off += (size_t)N * 64;
    float* q      = ws + off; off += (size_t)N * 64;
    float* kk     = ws + off; off += (size_t)N * 64;
    float* v      = ws + off; off += (size_t)N * 64;
    float* skip   = ws + off; off += (size_t)N * 64;
    float* agg    = ws + off; off += (size_t)N * 64;
    float* cc     = ws + off; off += (size_t)N * 72;
    float* segmax = ws + off; off += (size_t)N * 2;
    float* denom  = ws + off; off += (size_t)N * 2;
    float* score  = ws + off; off += (size_t)E * 2;

    float* out_h1 = (float*)d_out;                 // N*64
    float* out_ls = out_h1 + (size_t)N * 64;       // N*2
    float* out_y  = out_ls + (size_t)N * 2;        // N
    float* out_m  = out_y  + (size_t)N;            // N

    const int mtiles  = (N + 15) / 16;
    const int eblocks = (E + 127) / 128;

    init_buffers<<<dim3((N * 64 + 255) / 256), dim3(256), 0, stream>>>(segmax, denom, agg, N);
    gemm_n64_t<128, true><<<dim3(mtiles), dim3(128), 0, stream>>>(x, W_lin, b_lin, h, N);
    gemm_qkvs<<<dim3(mtiles), dim3(128), 0, stream>>>(h, Wq, bq, Wk, bk, Wv, bv,
                                                      Wskip, bskip, q, kk, v, skip, N);
    edge_score<<<dim3(eblocks), dim3(128), 0, stream>>>(ei, node_time, edge_time,
                                                        w_t, b_t, We, be, A1, a1, A2, a2,
                                                        q, kk, score, segmax, E);
    seg_softmax_norm<<<dim3((2 * E + 255) / 256), dim3(256), 0, stream>>>(ei, score, segmax,
                                                                          denom, E);
    edge_agg<<<dim3(eblocks), dim3(128), 0, stream>>>(ei, node_time, edge_time,
                                                      w_t, b_t, We, be, v, score, denom,
                                                      agg, E);
    node_fuse<<<dim3((N + 255) / 256), dim3(256), 0, stream>>>(agg, skip, tfi, degv,
                                                               w_tf, b_tf, w_deg, b_deg,
                                                               Wenc, benc, Wx, bx, cc, N);
    gemm_n64_t<72, false><<<dim3(mtiles), dim3(128), 0, stream>>>(cc, Wcomb, bcomb, out_h1, N);
    out_head<<<dim3((N + 255) / 256), dim3(256), 0, stream>>>(out_h1, Wout, bout, y, mask,
                                                              out_ls, out_y, out_m, N);
}
